// MoEGraphLayer_38371237822640
// MI455X (gfx1250) — matrix operand order repaired and checked by reading, not compiled, not run on hardware
//
#include <hip/hip_runtime.h>

// ---------------- problem constants (from reference setup_inputs) ----------
constexpr int BB   = 4;      // batch
constexpr int SS   = 1024;   // nodes
constexpr int DD   = 256;    // feature dim
constexpr int HH   = 6;      // heads
constexpr int HDIM = 64;     // head dim
constexpr int HDC  = HH * HDIM; // 384 concat dim
constexpr int NE   = 4;      // 1 main + 3 deputies
constexpr int BS   = BB * SS;   // 4096 rows
constexpr int SECT = 32, DOC = 4;

typedef __attribute__((ext_vector_type(16))) __bf16    v16bf;
typedef __attribute__((ext_vector_type(8)))  float     v8f;
typedef __attribute__((ext_vector_type(4)))  unsigned  v4u;
typedef __attribute__((ext_vector_type(8)))  unsigned  v8u;

__device__ __forceinline__ bool colmask(int e, int c) {
  if (e == 0) return true;                       // main: all columns
  if (e == 1) return c < SS - SECT - DOC;        // sentence cols
  if (e == 2) return c >= SS - SECT - DOC && c < SS - DOC; // section cols
  return c >= SS - DOC;                          // doc cols
}

// ---------------- Tensor Data Mover: 2D tile (bf16) global -> LDS ----------
// D# per CDNA5 ISA. data_size=2B, 2D tile,
// LDS padding enabled so consecutive rows are bank-staggered & 32B aligned.
// Issued by one wave; completion via TENSORcnt.
__device__ __forceinline__ void tdm_load_2d(const void* gptr, unsigned lds_addr,
                                            unsigned tensor_w, unsigned rows,
                                            unsigned cols, unsigned stride0,
                                            unsigned pad_int, unsigned pad_amt) {
  unsigned long long ga = (unsigned long long)gptr;
  union { unsigned long long q[2]; v4u v; } g0;
  // count=1 (valid), lds_addr at [63:32]
  g0.q[0] = 1ull | ((unsigned long long)lds_addr << 32);
  // global_addr at [120:64] (57b), type=2 at [127:126]
  g0.q[1] = (ga & 0x01FFFFFFFFFFFFFFull) | (2ull << 62);
  union { unsigned long long q[4]; v8u v; } g1;
  unsigned long long td0 = tensor_w;     // tensor_dim0
  unsigned long long td1 = rows;         // tensor_dim1
  unsigned long long tl0 = cols;         // tile_dim0
  unsigned long long tl1 = rows;         // tile_dim1
  unsigned long long st0 = stride0;      // tensor_dim0_stride
  // [17:16]=data_size(1->2B), [20]=pad_enable, [24:22]=pad_interval,
  // [31:25]=pad_amount, [79:48]=tensor_dim0 (low 16 in q0)
  g1.q[0] = (1ull << 16) | (1ull << 20)
          | ((unsigned long long)(pad_int & 7u) << 22)
          | ((unsigned long long)(pad_amt & 127u) << 25)
          | ((td0 & 0xffffull) << 48);
  // [79:64]=tensor_dim0 hi, [111:80]=tensor_dim1, [127:112]=tile_dim0
  g1.q[1] = ((td0 >> 16) & 0xffffull) | ((td1 & 0xffffffffull) << 16)
          | ((tl0 & 0xffffull) << 48);
  // [143:128]=tile_dim1, [159:144]=tile_dim2(0), [207:160]=tensor_dim0_stride
  g1.q[2] = (tl1 & 0xffffull) | ((st0 & 0xffffffffull) << 32);
  // [207:192]=stride hi16, [255:208]=tensor_dim1_stride (unused for 2D)
  g1.q[3] = ((st0 >> 32) & 0xffffull);
  asm volatile("tensor_load_to_lds %0, %1" :: "s"(g0.v), "s"(g1.v) : "memory");
}

// ---------------- 1. router + masked bf16 feature copies -------------------
__global__ void prep_kernel(const float* __restrict__ x, const float* __restrict__ rW,
                            __bf16* __restrict__ xbf, float* __restrict__ maskp) {
  int idx = blockIdx.x * blockDim.x + threadIdx.x;   // (b,s) row, 0..BS
  const float* xr = x + (size_t)idx * DD;
  float l0 = 0.f, l1 = 0.f, l2 = 0.f;
  for (int f = 0; f < DD; ++f) {
    float xv = xr[f];
    l0 += xv * rW[f * 3 + 0];
    l1 += xv * rW[f * 3 + 1];
    l2 += xv * rW[f * 3 + 2];
  }
  int amin = 0; float mv = l0;
  if (l1 < mv) { mv = l1; amin = 1; }
  if (l2 < mv) { mv = l2; amin = 2; }
  float mk[NE];
  mk[0] = 1.f;                        // main always on
  mk[1] = (amin == 0) ? 0.f : 1.f;    // top-2 of 3 == all but argmin
  mk[2] = (amin == 1) ? 0.f : 1.f;
  mk[3] = (amin == 2) ? 0.f : 1.f;
  for (int e = 0; e < NE; ++e) maskp[e * BS + idx] = mk[e];
  for (int f = 0; f < DD; ++f) {
    float xv = xr[f];
    for (int e = 0; e < NE; ++e)
      xbf[(size_t)e * BS * DD + (size_t)idx * DD + f] = (__bf16)(xv * mk[e]);
  }
}

// ---------------- 2. weight conversion ------------------------------------
__global__ void convert_w1(const float* __restrict__ mw, const float* __restrict__ dw,
                           __bf16* __restrict__ out) {
  int idx = blockIdx.x * blockDim.x + threadIdx.x;   // NE*DD*HDC
  int e = idx / (DD * HDC);
  int r = idx % (DD * HDC);
  int f = r / HDC, c = r % HDC;
  int h = c / HDIM, o = c % HDIM;
  float v = (e == 0) ? mw[(h * DD + f) * HDIM + o]
                     : dw[(((e - 1) * HH + h) * DD + f) * HDIM + o];
  out[idx] = (__bf16)v;
}

__global__ void convert_w2(const float* __restrict__ mw, const float* __restrict__ dw,
                           __bf16* __restrict__ out) {
  int idx = blockIdx.x * blockDim.x + threadIdx.x;   // NE*HDC*DD
  int e = idx / (HDC * DD);
  int r = idx % (HDC * DD);
  float v = (e == 0) ? mw[r] : dw[(e - 1) * HDC * DD + r];
  out[idx] = (__bf16)v;
}

__global__ void convert_copy(const float* __restrict__ src, __bf16* __restrict__ dst, int n) {
  int idx = blockIdx.x * blockDim.x + threadIdx.x;
  if (idx < n) dst[idx] = (__bf16)src[idx];
}

// ---------------- 3. WMMA GEMM with TDM-staged weights ---------------------
// block = 256 thr = 8 waves; block tile = 16 rows x 128 cols.
// Weight tile [32 x 128] double-buffered in LDS via tensor_load_to_lds.
// LDS pad: 128 elems = 64 DW -> pad_interval=5; pad 8 DW -> row stride 144.
__global__ void gemm_bf16(const __bf16* __restrict__ A, const __bf16* __restrict__ W,
                          float* __restrict__ Cf, __bf16* __restrict__ Cb,
                          int lda, int ldb, int ldc, int K,
                          size_t sA, size_t sB, size_t sC) {
  __shared__ __align__(128) __bf16 wlds[2][32 * 144];
  int wave = threadIdx.x >> 5, lane = threadIdx.x & 31;
  int m0 = blockIdx.x * 16;
  int nblk = blockIdx.y * 128;
  const __bf16* Ab = A + (size_t)blockIdx.z * sA;
  const __bf16* Wb = W + (size_t)blockIdx.z * sB + nblk;
  size_t cbase = (size_t)blockIdx.z * sC;
  const __bf16* Arow = Ab + (size_t)(m0 + (lane & 15)) * lda + (lane >> 4) * 16;

  if (wave == 0) {
    tdm_load_2d(Wb, (unsigned)(unsigned long long)&wlds[0][0],
                (unsigned)ldb, 32, 128, (unsigned)ldb, 5, 7);
    __builtin_amdgcn_s_wait_tensorcnt(0);
  }
  __syncthreads();

  v8f acc = {};
  for (int kb = 0; kb < K; kb += 32) {
    int buf = (kb >> 5) & 1;
    if (wave == 0 && kb + 32 < K)
      tdm_load_2d(Wb + (size_t)(kb + 32) * ldb,
                  (unsigned)(unsigned long long)&wlds[buf ^ 1][0],
                  (unsigned)ldb, 32, 128, (unsigned)ldb, 5, 7);
    v16bf a = *(const v16bf*)(Arow + kb);
    __builtin_prefetch(Arow + kb + 64, 0, 0);   // global_prefetch_b8
    v16bf b;
    for (int s = 0; s < 16; ++s)
      b[s] = wlds[buf][((lane >> 4) * 16 + s) * 144 + wave * 16 + (lane & 15)];
    acc = __builtin_amdgcn_wmma_f32_16x16x32_bf16(false, a, false, b,
                                                  (short)0, acc, false, false);
    if (wave == 0) __builtin_amdgcn_s_wait_tensorcnt(0);
    __syncthreads();
  }
  for (int v = 0; v < 8; ++v) {
    int mr = m0 + (lane >> 4) * 8 + v;
    size_t idx = cbase + (size_t)mr * ldc + nblk + wave * 16 + (lane & 15);
    if (Cf) Cf[idx] = acc[v];
    else    Cb[idx] = (__bf16)acc[v];
  }
}

// ---------------- 4. transpose [R,C] -> [C,R] per batch (grid.y) -----------
__global__ void transpose_bf16(const __bf16* __restrict__ in, __bf16* __restrict__ out,
                               int R, int C) {
  size_t idx = (size_t)blockIdx.x * blockDim.x + threadIdx.x;
  if (idx >= (size_t)R * C) return;
  int batch = blockIdx.y;
  int r = (int)(idx / C), c = (int)(idx % C);
  out[((size_t)batch * C + c) * R + r] = in[((size_t)batch * R + r) * C + c];
}

// ---------------- 5. es/ed row dots (layer-1 attention logits) -------------
__global__ void rowdot_kernel(const __bf16* __restrict__ h,
                              const float* __restrict__ ma1s, const float* __restrict__ ma1d,
                              const float* __restrict__ da1s, const float* __restrict__ da1d,
                              float* __restrict__ es, float* __restrict__ ed) {
  int idx = blockIdx.x * blockDim.x + threadIdx.x;  // ((e*BB+b)*HH+hh)*SS+s
  int s = idx % SS; int t = idx / SS;
  int hh = t % HH; t /= HH;
  int b = t % BB; int e = t / BB;
  const __bf16* hrow = h + ((size_t)(e * BB + b) * SS + s) * HDC + hh * HDIM;
  const float* a1s = (e == 0) ? ma1s + hh * HDIM : da1s + ((e - 1) * HH + hh) * HDIM;
  const float* a1d = (e == 0) ? ma1d + hh * HDIM : da1d + ((e - 1) * HH + hh) * HDIM;
  float as = 0.f, ad = 0.f;
  for (int o = 0; o < HDIM; ++o) {
    float hv = (float)hrow[o];
    as += hv * a1s[o];
    ad += hv * a1d[o];
  }
  es[idx] = as; ed[idx] = ad;
}

// ---------------- 6. softmax stats: online max/sum, wave per row -----------
__global__ void stats_kernel(const float* __restrict__ es, const float* __restrict__ ed,
                             const float* __restrict__ adj,
                             float* __restrict__ mo, float* __restrict__ so, int Hpar) {
  int wave = threadIdx.x >> 5, lane = threadIdx.x & 31;
  int row = blockIdx.x * 4 + wave;
  int s = row % SS; int t = row / SS;
  int hh = t % Hpar; t /= Hpar;
  int b = t % BB; int e = t / BB;
  const float* adjrow = adj + ((size_t)b * SS + s) * SS;
  const float* edrow = ed + (size_t)((e * BB + b) * Hpar + hh) * SS;
  float esr = es[row];
  float m = -1e30f, sum = 0.f;
  for (int c = lane; c < SS; c += 32) {
    if (adjrow[c] > 0.f && colmask(e, c)) {
      float v = esr + edrow[c];
      v = v > 0.f ? v : 0.2f * v;         // leaky_relu(0.2)
      float mn = fmaxf(m, v);
      sum = sum * __expf(m - mn) + __expf(v - mn);
      m = mn;
    }
  }
  for (int off = 16; off; off >>= 1) {
    float m2 = __shfl_xor(m, off, 32);
    float s2 = __shfl_xor(sum, off, 32);
    float mn = fmaxf(m, m2);
    sum = sum * __expf(m - mn) + s2 * __expf(m2 - mn);
    m = mn;
  }
  if (lane == 0) { mo[row] = m; so[row] = sum; }
}

// ---------------- 7. attention-1 apply: h1 = elu(att @ h) ------------------
// grid: (SS/16, NE*BB*HH), 128 thr = 4 waves; wave w -> n-tile w (Hd=64).
// hT tile [64 x 32] double-buffered in LDS via TDM (pad -> row stride 48).
__global__ void att1_apply(const __bf16* __restrict__ hT,
                           const float* __restrict__ es, const float* __restrict__ ed,
                           const float* __restrict__ m1, const float* __restrict__ s1,
                           const float* __restrict__ adj, __bf16* __restrict__ h1) {
  __shared__ __align__(64) __bf16 attT[16][32];
  __shared__ __align__(128) __bf16 stage[2][64 * 48];
  __shared__ float mrow[16], invs[16];
  int tid = threadIdx.x, wave = tid >> 5, lane = tid & 31;
  int y = blockIdx.y;
  int hh = y % HH, b = (y / HH) % BB, e = y / (HH * BB);
  int row0 = blockIdx.x * 16;
  size_t statbase = ((size_t)(e * BB + b) * HH + hh) * SS;
  const __bf16* gB = hT + (size_t)((e * BB + b) * HDC + hh * HDIM) * SS;
  if (tid < 16) {
    float mm = m1[statbase + row0 + tid];
    float sm = s1[statbase + row0 + tid];
    mrow[tid] = mm;
    invs[tid] = sm > 1e-30f ? 1.f / sm : 0.f;
  }
  if (wave == 0) {
    tdm_load_2d(gB, (unsigned)(unsigned long long)&stage[0][0],
                SS, 64, 32, SS, 3, 7);
    __builtin_amdgcn_s_wait_tensorcnt(0);
  }
  __syncthreads();
  const float* adjb = adj + (size_t)b * SS * SS;
  const float* edrow = ed + statbase;
  v8f acc = {};
  for (int kb = 0; kb < SS; kb += 32) {
    int buf = (kb >> 5) & 1;
    if (wave == 0 && kb + 32 < SS)
      tdm_load_2d(gB + kb + 32, (unsigned)(unsigned long long)&stage[buf ^ 1][0],
                  SS, 64, 32, SS, 3, 7);
    for (int t = tid; t < 512; t += 128) {        // 16x32 prob tile
      int rr = t >> 5, cc = t & 31;
      int c = kb + cc, sg = row0 + rr;
      float p = 0.f;
      if (adjb[(size_t)sg * SS + c] > 0.f && colmask(e, c)) {
        float v = es[statbase + sg] + edrow[c];
        v = v > 0.f ? v : 0.2f * v;
        p = __expf(v - mrow[rr]) * invs[rr];
      }
      attT[rr][cc] = (__bf16)p;
    }
    __syncthreads();
    v16bf a = *(const v16bf*)&attT[lane & 15][(lane >> 4) * 16];
    v16bf bfrag = *(const v16bf*)&stage[buf][(wave * 16 + (lane & 15)) * 48
                                            + (lane >> 4) * 16];
    acc = __builtin_amdgcn_wmma_f32_16x16x32_bf16(false, a, false, bfrag,
                                                  (short)0, acc, false, false);
    if (wave == 0) __builtin_amdgcn_s_wait_tensorcnt(0);
    __syncthreads();
  }
  for (int v = 0; v < 8; ++v) {
    int sg = row0 + (lane >> 4) * 8 + v;
    float val = acc[v];
    val = val > 0.f ? val : __expf(val) - 1.f;    // ELU
    h1[((size_t)(e * BB + b) * SS + sg) * HDC + hh * HDIM + wave * 16 + (lane & 15)]
        = (__bf16)val;
  }
}

// ---------------- 8. s2/d2 row dots for layer-2 attention ------------------
__global__ void sd2_kernel(const __bf16* __restrict__ h2,
                           const float* __restrict__ ma2s, const float* __restrict__ ma2d,
                           const float* __restrict__ da2s, const float* __restrict__ da2d,
                           float* __restrict__ s2, float* __restrict__ d2) {
  int idx = blockIdx.x * blockDim.x + threadIdx.x;   // (e*BB+b)*SS+s
  int s = idx % SS; int t = idx / SS;
  int b = t % BB; int e = t / BB;
  const __bf16* row = h2 + ((size_t)(e * BB + b) * SS + s) * DD;
  const float* a2s = (e == 0) ? ma2s : da2s + (e - 1) * DD;
  const float* a2d = (e == 0) ? ma2d : da2d + (e - 1) * DD;
  float as = 0.f, ad = 0.f;
  for (int o = 0; o < DD; ++o) {
    float hv = (float)row[o];
    as += hv * a2s[o];
    ad += hv * a2d[o];
  }
  s2[idx] = as; d2[idx] = ad;
}

// ---------------- 9. attention-2 apply: out = att2 @ h2 --------------------
// grid: (SS/16, NE*BB, 4); wave w -> n-tile blockIdx.z*4+w (D=256).
// h2T tile [64 x 32] double-buffered in LDS via TDM.
__global__ void att2_apply(const __bf16* __restrict__ h2T,
                           const float* __restrict__ s2, const float* __restrict__ d2,
                           const float* __restrict__ m2, const float* __restrict__ z2,
                           const float* __restrict__ adj, const float* __restrict__ maskp,
                           float* __restrict__ main_out, float* __restrict__ dep_acc) {
  __shared__ __align__(64) __bf16 attT[16][32];
  __shared__ __align__(128) __bf16 stage[2][64 * 48];
  __shared__ float mrow[16], invs[16];
  int tid = threadIdx.x, wave = tid >> 5, lane = tid & 31;
  int y = blockIdx.y;
  int b = y % BB, e = y / BB;
  int row0 = blockIdx.x * 16;
  size_t statbase = (size_t)(e * BB + b) * SS;
  const __bf16* gB = h2T + (size_t)((e * BB + b) * DD + blockIdx.z * 64) * SS;
  if (tid < 16) {
    float mm = m2[statbase + row0 + tid];
    float sm = z2[statbase + row0 + tid];
    mrow[tid] = mm;
    invs[tid] = sm > 1e-30f ? 1.f / sm : 0.f;
  }
  if (wave == 0) {
    tdm_load_2d(gB, (unsigned)(unsigned long long)&stage[0][0],
                SS, 64, 32, SS, 3, 7);
    __builtin_amdgcn_s_wait_tensorcnt(0);
  }
  __syncthreads();
  const float* adjb = adj + (size_t)b * SS * SS;
  const float* d2row = d2 + statbase;
  v8f acc = {};
  for (int kb = 0; kb < SS; kb += 32) {
    int buf = (kb >> 5) & 1;
    if (wave == 0 && kb + 32 < SS)
      tdm_load_2d(gB + kb + 32, (unsigned)(unsigned long long)&stage[buf ^ 1][0],
                  SS, 64, 32, SS, 3, 7);
    for (int t = tid; t < 512; t += 128) {
      int rr = t >> 5, cc = t & 31;
      int c = kb + cc, sg = row0 + rr;
      float p = 0.f;
      if (adjb[(size_t)sg * SS + c] > 0.f && colmask(e, c)) {
        float v = s2[statbase + sg] + d2row[c];
        v = v > 0.f ? v : 0.2f * v;
        p = __expf(v - mrow[rr]) * invs[rr];
      }
      attT[rr][cc] = (__bf16)p;
    }
    __syncthreads();
    v16bf a = *(const v16bf*)&attT[lane & 15][(lane >> 4) * 16];
    v16bf bfrag = *(const v16bf*)&stage[buf][(wave * 16 + (lane & 15)) * 48
                                            + (lane >> 4) * 16];
    acc = __builtin_amdgcn_wmma_f32_16x16x32_bf16(false, a, false, bfrag,
                                                  (short)0, acc, false, false);
    if (wave == 0) __builtin_amdgcn_s_wait_tensorcnt(0);
    __syncthreads();
  }
  for (int v = 0; v < 8; ++v) {
    int sg = row0 + (lane >> 4) * 8 + v;
    int n = blockIdx.z * 64 + wave * 16 + (lane & 15);
    size_t oidx = ((size_t)b * SS + sg) * DD + n;
    float val = acc[v];
    if (e == 0) main_out[oidx] = val;
    else atomicAdd(&dep_acc[oidx], val * maskp[e * BS + b * SS + sg]);
  }
}

// ---------------- 10. init / finalize --------------------------------------
__global__ void init_kernel(float* __restrict__ dep_acc, float* __restrict__ scal, int n) {
  int idx = blockIdx.x * blockDim.x + threadIdx.x;
  if (idx < n) dep_acc[idx] = 0.f;
  if (idx == 0) scal[0] = 0.f;
}

__global__ void finalize_kernel(const float* __restrict__ logits, const float* __restrict__ bias,
                                const float* __restrict__ mo, const float* __restrict__ da,
                                float* __restrict__ out, float* __restrict__ scal) {
  int idx = blockIdx.x * blockDim.x + threadIdx.x;   // BS*DD
  float l = logits[idx] + bias[idx % DD];
  float sig = 1.f / (1.f + __expf(-l));
  out[idx] = sig * mo[idx] + (1.f - sig) * da[idx];
  float v = sig;
  for (int off = 16; off; off >>= 1) v += __shfl_xor(v, off, 32);
  if ((threadIdx.x & 31) == 0) atomicAdd(scal, v);
}

__global__ void scalar_kernel(const float* __restrict__ scal, float* __restrict__ out) {
  float mc = scal[0] / (float)(BS * DD);
  out[BS * DD]     = fabsf(mc - 0.6f) * 0.01f;   // contribution_loss
  out[BS * DD + 1] = mc;                          // main_contribution
}

// ---------------- launch ---------------------------------------------------
extern "C" void kernel_launch(void* const* d_in, const int* in_sizes, int n_in,
                              void* d_out, int out_size, void* d_ws, size_t ws_size,
                              hipStream_t stream) {
  const float* feature  = (const float*)d_in[0];
  const float* adj      = (const float*)d_in[1];
  const float* main_W1  = (const float*)d_in[2];
  const float* main_a1s = (const float*)d_in[3];
  const float* main_a1d = (const float*)d_in[4];
  const float* main_W2  = (const float*)d_in[5];
  const float* main_a2s = (const float*)d_in[6];
  const float* main_a2d = (const float*)d_in[7];
  const float* dep_W1   = (const float*)d_in[8];
  const float* dep_a1s  = (const float*)d_in[9];
  const float* dep_a1d  = (const float*)d_in[10];
  const float* dep_W2   = (const float*)d_in[11];
  const float* dep_a2s  = (const float*)d_in[12];
  const float* dep_a2d  = (const float*)d_in[13];
  const float* router_W = (const float*)d_in[14];
  const float* blend_W  = (const float*)d_in[15];
  const float* blend_b  = (const float*)d_in[16];

  char* ws = (char*)d_ws;
  size_t off = 0;
  auto alloc = [&](size_t bytes) -> char* {
    char* p = ws + off;
    off = (off + bytes + 255) & ~(size_t)255;
    return p;
  };

  __bf16* xbf   = (__bf16*)alloc((size_t)NE * BS * DD * 2);
  __bf16* w1    = (__bf16*)alloc((size_t)NE * DD * HDC * 2);
  __bf16* w2    = (__bf16*)alloc((size_t)NE * HDC * DD * 2);
  __bf16* bw    = (__bf16*)alloc((size_t)DD * DD * 2);
  float*  maskp = (float*) alloc((size_t)NE * BS * 4);
  __bf16* hbuf  = (__bf16*)alloc((size_t)NE * BS * HDC * 2);
  __bf16* hT    = (__bf16*)alloc((size_t)NE * BS * HDC * 2);
  float*  es    = (float*) alloc((size_t)NE * BB * HH * SS * 4);
  float*  ed    = (float*) alloc((size_t)NE * BB * HH * SS * 4);
  float*  m1    = (float*) alloc((size_t)NE * BB * HH * SS * 4);
  float*  s1    = (float*) alloc((size_t)NE * BB * HH * SS * 4);
  __bf16* h1    = (__bf16*)alloc((size_t)NE * BS * HDC * 2);
  __bf16* h2    = (__bf16*)alloc((size_t)NE * BS * DD * 2);
  __bf16* h2T   = (__bf16*)alloc((size_t)NE * BS * DD * 2);
  float*  s2v   = (float*) alloc((size_t)NE * BS * 4);
  float*  d2v   = (float*) alloc((size_t)NE * BS * 4);
  float*  m2    = (float*) alloc((size_t)NE * BS * 4);
  float*  z2    = (float*) alloc((size_t)NE * BS * 4);
  float*  main_out = (float*)alloc((size_t)BS * DD * 4);
  float*  dep_acc  = (float*)alloc((size_t)BS * DD * 4);
  float*  blendL   = (float*)alloc((size_t)BS * DD * 4);
  float*  scal     = (float*)alloc(256);

  float* out = (float*)d_out;

  prep_kernel<<<BS / 128, 128, 0, stream>>>(feature, router_W, xbf, maskp);
  convert_w1<<<(NE * DD * HDC) / 256, 256, 0, stream>>>(main_W1, dep_W1, w1);
  convert_w2<<<(NE * HDC * DD) / 256, 256, 0, stream>>>(main_W2, dep_W2, w2);
  convert_copy<<<(DD * DD) / 256, 256, 0, stream>>>(blend_W, bw, DD * DD);
  init_kernel<<<(BS * DD) / 256, 256, 0, stream>>>(dep_acc, scal, BS * DD);

  // h = x_e @ W1cat : [4096,256]x[256,384] per expert  (WMMA + TDM weights)
  gemm_bf16<<<dim3(BS / 16, 3, NE), 256, 0, stream>>>(
      xbf, w1, nullptr, hbuf, DD, HDC, HDC, DD,
      (size_t)BS * DD, (size_t)DD * HDC, (size_t)BS * HDC);
  transpose_bf16<<<dim3((SS * HDC) / 256, NE * BB), 256, 0, stream>>>(hbuf, hT, SS, HDC);

  rowdot_kernel<<<(NE * BB * HH * SS) / 256, 256, 0, stream>>>(
      hbuf, main_a1s, main_a1d, dep_a1s, dep_a1d, es, ed);
  stats_kernel<<<(NE * BB * HH * SS) / 4, 128, 0, stream>>>(es, ed, adj, m1, s1, HH);
  att1_apply<<<dim3(SS / 16, NE * BB * HH), 128, 0, stream>>>(hT, es, ed, m1, s1, adj, h1);

  // h2 = h1 @ W2 : [4096,384]x[384,256] per expert  (WMMA + TDM weights)
  gemm_bf16<<<dim3(BS / 16, 2, NE), 256, 0, stream>>>(
      h1, w2, nullptr, h2, HDC, DD, DD, HDC,
      (size_t)BS * HDC, (size_t)HDC * DD, (size_t)BS * DD);
  transpose_bf16<<<dim3((SS * DD) / 256, NE * BB), 256, 0, stream>>>(h2, h2T, SS, DD);

  sd2_kernel<<<(NE * BS) / 256, 256, 0, stream>>>(
      h2, main_a2s, main_a2d, dep_a2s, dep_a2d, s2v, d2v);
  stats_kernel<<<(NE * BS) / 4, 128, 0, stream>>>(s2v, d2v, adj, m2, z2, 1);
  att2_apply<<<dim3(SS / 16, NE * BB, 4), 128, 0, stream>>>(
      h2T, s2v, d2v, m2, z2, adj, maskp, main_out, dep_acc);

  gemm_bf16<<<dim3(BS / 16, 2, 1), 256, 0, stream>>>(
      xbf, bw, blendL, nullptr, DD, DD, DD, DD, 0, 0, 0);
  finalize_kernel<<<(BS * DD) / 256, 256, 0, stream>>>(
      blendL, blend_b, main_out, dep_acc, out, scal);
  scalar_kernel<<<1, 1, 0, stream>>>(scal, out);
}
